// CustomLSTM_10325101379619
// MI455X (gfx1250) — matrix operand order repaired and checked by reading, not compiled, not run on hardware
//
#include <hip/hip_runtime.h>
#include <hip/hip_bf16.h>

// ---------------------------------------------------------------------------
// LSTM: B=64, S=512, IN=1024, H=1024.
//   Phase 1: xW[s][b][4H] = x[b][s][:] @ W + bias       (WMMA bf16 GEMM)
//   Phase 2: persistent kernel, 512 serial steps:
//              gates = xW[t] + h @ U  -> i,f,g,o -> c,h
//            h staged per-step into LDS via async-to-LDS double buffering.
// Mixed precision: bf16 operands, fp32 accumulate/state.
// ---------------------------------------------------------------------------

#define BS   64
#define SEQ  512
#define HS   1024
#define G4   4096   // 4*HS
#define KC   128    // K-chunk staged in LDS per pipeline stage

typedef __bf16          v16bf __attribute__((ext_vector_type(16)));
typedef unsigned short  v16u  __attribute__((ext_vector_type(16)));
typedef unsigned short  v8u   __attribute__((ext_vector_type(8)));
typedef float           v8f   __attribute__((ext_vector_type(8)));
typedef int             v4i   __attribute__((ext_vector_type(4)));

typedef __attribute__((address_space(3))) unsigned short lds_u16;
typedef __attribute__((address_space(3))) v8u            lds_v8u;

// Workspace layout (bytes)
#define OFF_XW   ((size_t)0)                      // fp32 xW: 512*64*4096*4 = 512MB
#define OFF_WT   (OFF_XW + (size_t)SEQ*BS*G4*4)   // bf16 W^T [4096][1024] = 8MB
#define OFF_UT   (OFF_WT + (size_t)G4*HS*2)       // bf16 U^T [4096][1024] = 8MB
#define OFF_XB   (OFF_UT + (size_t)G4*HS*2)       // bf16 x   [64][512][1024] = 64MB
#define OFF_HB   (OFF_XB + (size_t)BS*SEQ*HS*2)   // bf16 h, double-buffered: 256KB
#define OFF_C    (OFF_HB + (size_t)2*BS*HS*2)     // fp32 c: 256KB
#define OFF_BAR  (OFF_C  + (size_t)BS*HS*4)       // u32 barrier counters: 512*4

#if defined(__gfx1250__) && __has_builtin(__builtin_amdgcn_global_load_async_to_lds_b128)
#define HAVE_ASYNC_LDS 1
#else
#define HAVE_ASYNC_LDS 0
#endif

__device__ __forceinline__ unsigned short f2bf(float f) {
    union { float f; unsigned u; } v; v.f = f;
    unsigned u = v.u;
    return (unsigned short)((u + 0x7FFFu + ((u >> 16) & 1u)) >> 16);
}

__device__ __forceinline__ float sigm(float x) {
    return 1.0f / (1.0f + __expf(-x));
}

// Copy 16 bytes global->LDS. Async path uses GLOBAL_LOAD_ASYNC_TO_LDS_B128
// (tracked by ASYNCcnt); fallback is a synchronous generic copy.
// The builtin wants (int4 addrspace(1)*, int4 addrspace(3)*, imm, imm).
// Generic->LDS pointer: flat LDS aperture maps addr[31:0] to the LDS offset,
// so truncating the flat address yields a valid addrspace(3) pointer value.
__device__ __forceinline__ void cp16_g2l(const void* g, void* l) {
#if HAVE_ASYNC_LDS
    __builtin_amdgcn_global_load_async_to_lds_b128(
        (__attribute__((address_space(1))) v4i*)(unsigned long long)(size_t)g,
        (__attribute__((address_space(3))) v4i*)(unsigned int)(size_t)l,
        0, 0);
#else
    *(v8u*)l = *(const v8u*)g;
#endif
}

__device__ __forceinline__ void wait_async_le4() {
#if HAVE_ASYNC_LDS
#if __has_builtin(__builtin_amdgcn_s_wait_asynccnt)
    __builtin_amdgcn_s_wait_asynccnt(4);
#else
    asm volatile("s_wait_asynccnt 0x4" ::: "memory");
#endif
#endif
}

__device__ __forceinline__ void wait_async_le0() {
#if HAVE_ASYNC_LDS
#if __has_builtin(__builtin_amdgcn_s_wait_asynccnt)
    __builtin_amdgcn_s_wait_asynccnt(0);
#else
    asm volatile("s_wait_asynccnt 0x0" ::: "memory");
#endif
#endif
}

// A fragment (16x32, bf16, row-major global source, ld elements):
// lane half 0: K 0..15; half 1: K 16..31; row = tile_m0 + lane%16
__device__ __forceinline__ v16bf load_a(const unsigned short* A, int ld,
                                        int row, int k, int half) {
    const v8u* p0 = (const v8u*)(A + (size_t)row * ld + k + half * 16);
    const v8u* p1 = (const v8u*)(A + (size_t)row * ld + k + half * 16 + 8);
    v8u a0 = *p0, a1 = *p1;
    v16u r;
#pragma unroll
    for (int i = 0; i < 8; ++i) { r[i] = a0[i]; r[8 + i] = a1[i]; }
    return __builtin_bit_cast(v16bf, r);
}

// Same, but from an LDS-resident tile (addrspace(3) -> ds_load_b128)
__device__ __forceinline__ v16bf load_a_lds(const lds_u16* A, int ld,
                                            int row, int k, int half) {
    const lds_v8u* p0 = (const lds_v8u*)(A + row * ld + k + half * 16);
    const lds_v8u* p1 = (const lds_v8u*)(A + row * ld + k + half * 16 + 8);
    v8u a0 = *p0, a1 = *p1;
    v16u r;
#pragma unroll
    for (int i = 0; i < 8; ++i) { r[i] = a0[i]; r[8 + i] = a1[i]; }
    return __builtin_bit_cast(v16bf, r);
}

// B fragment (32x16, bf16) from N-major (transposed) storage Bt[N][K], ld = K:
// lane = column N (mod 16); half 0: K 0..15, half 1: K 16..31 -> one 32B load
__device__ __forceinline__ v16bf load_b(const unsigned short* Bt, int ld,
                                        int col, int k, int half) {
    const v16u* p = (const v16u*)(Bt + (size_t)col * ld + k + half * 16);
    return __builtin_bit_cast(v16bf, *p);
}

__device__ __forceinline__ v8f wmma_bf16(v16bf a, v16bf b, v8f c) {
    return __builtin_amdgcn_wmma_f32_16x16x32_bf16(false, a, false, b,
                                                   (short)0, c, false, false);
}

// ---------------------------------------------------------------------------
// Init: zero h (both buffers), c, barrier counters.  Runs every launch so
// graph replay starts from a clean state.
// ---------------------------------------------------------------------------
__global__ void k_init(unsigned short* hb, float* c, unsigned int* bar) {
    int stride = gridDim.x * blockDim.x;
    int i = blockIdx.x * blockDim.x + threadIdx.x;
    for (int j = i; j < 2 * BS * HS; j += stride) hb[j] = 0;
    for (int j = i; j < BS * HS;     j += stride) c[j]  = 0.0f;
    for (int j = i; j < SEQ;         j += stride) bar[j] = 0u;
}

// fp32 -> bf16 bulk convert (x)
__global__ void k_f2bf(const float* __restrict__ in, unsigned short* __restrict__ out,
                       size_t n) {
    size_t stride = (size_t)gridDim.x * blockDim.x;
    for (size_t j = (size_t)blockIdx.x * blockDim.x + threadIdx.x; j < n; j += stride)
        out[j] = f2bf(in[j]);
}

// Transpose [HS][G4] fp32 -> [G4][HS] bf16 (one-time, for W and U)
__global__ void k_transpose_bf(const float* __restrict__ in,
                               unsigned short* __restrict__ out) {
    size_t total = (size_t)G4 * HS;
    size_t stride = (size_t)gridDim.x * blockDim.x;
    for (size_t idx = (size_t)blockIdx.x * blockDim.x + threadIdx.x; idx < total;
         idx += stride) {
        size_t n = idx >> 10;
        size_t k = idx & 1023;
        out[idx] = f2bf(in[k * G4 + n]);
    }
}

// ---------------------------------------------------------------------------
// Phase 1: xW[s][b][g] = x[b][s][:] @ W[:,g] + bias[g]
// Block = 64x64 tile (M = b*512+s flattened, N = g), 4 waves of 32x32.
// ---------------------------------------------------------------------------
__global__ __launch_bounds__(128)
void k_gemm_xw(const unsigned short* __restrict__ xb,
               const unsigned short* __restrict__ Wt,
               const float* __restrict__ bias,
               float* __restrict__ xW) {
    const int wave = threadIdx.x >> 5;
    const int lane = threadIdx.x & 31;
    const int half = lane >> 4;
    const int r    = lane & 15;
    const int m0 = blockIdx.y * 64 + (wave >> 1) * 32;   // flattened (b,s) row
    const int n0 = blockIdx.x * 64 + (wave & 1) * 32;    // gate column

    v8f acc[2][2] = {};
#pragma unroll 4
    for (int k = 0; k < HS; k += 32) {
        v16bf a0 = load_a(xb, HS, m0 + r,      k, half);
        v16bf a1 = load_a(xb, HS, m0 + 16 + r, k, half);
        v16bf b0 = load_b(Wt, HS, n0 + r,      k, half);
        v16bf b1 = load_b(Wt, HS, n0 + 16 + r, k, half);
        acc[0][0] = wmma_bf16(a0, b0, acc[0][0]);
        acc[0][1] = wmma_bf16(a0, b1, acc[0][1]);
        acc[1][0] = wmma_bf16(a1, b0, acc[1][0]);
        acc[1][1] = wmma_bf16(a1, b1, acc[1][1]);
    }

#pragma unroll
    for (int mi = 0; mi < 2; ++mi) {
#pragma unroll
        for (int ni = 0; ni < 2; ++ni) {
            int col = n0 + ni * 16 + r;
            float bv = bias[col];
#pragma unroll
            for (int v = 0; v < 8; ++v) {
                int row = m0 + mi * 16 + half * 8 + v;   // = b*512 + s
                int b   = row >> 9;
                int s   = row & 511;
                xW[(((size_t)s * BS + b) << 12) + col] = acc[mi][ni][v] + bv;
            }
        }
    }
}

// ---------------------------------------------------------------------------
// Phase 2: persistent recurrent kernel.  64 blocks; block owns 16 hidden
// columns, computes i/f/g/o tiles for them each step.  h_{t-1} is staged
// into LDS in double-buffered K-chunks via async-to-LDS copies; WMMA reads
// A from LDS (ds_load_b128) and B (U^T) from L2-resident global.
// ---------------------------------------------------------------------------
__global__ __launch_bounds__(256)
void k_recurrent(const unsigned short* __restrict__ Ut,
                 const float* __restrict__ xW,
                 unsigned short* __restrict__ hb,   // [2][BS][HS] bf16
                 float* __restrict__ cbuf,          // [BS][HS] fp32
                 float* __restrict__ out,           // hidden_seq | h_t | c_t
                 unsigned int* __restrict__ bar) {
    __shared__ unsigned short sh_h[2][BS][KC];   // 32KB: double-buffered h chunk
    __shared__ float          sh_g[4][BS][16];   // 16KB: gate pre-activations

    const int wave = threadIdx.x >> 5;
    const int lane = threadIdx.x & 31;
    const int half = lane >> 4;
    const int r    = lane & 15;
    const int q    = wave >> 1;          // gate: 0=i 1=f 2=g 3=o
    const int mh   = wave & 1;           // batch-row half (0..31 / 32..63)
    const int h0   = blockIdx.x * 16;    // hidden column tile base
    const int bcol = q * HS + h0 + r;    // this lane's gate column

    const int NCH = HS / KC;             // 8 chunks per step

    for (int t = 0; t < SEQ; ++t) {
        const unsigned short* hprev = hb + (size_t)(t & 1) * BS * HS;

        // Prologue: stage chunk 0 into buffer 0 (4 async b128 per wave)
#pragma unroll
        for (int i = 0; i < 4; ++i) {
            int o   = (int)threadIdx.x * 16 + i * 4096;   // byte offset in chunk
            int row = o >> 8;                             // KC*2 = 256B per row
            int rem = o & 255;
            cp16_g2l((const char*)hprev + row * (HS * 2) + rem,
                     (char*)&sh_h[0][0][0] + o);
        }

        v8f acc0 = {}, acc1 = {};
        for (int c = 0; c < NCH; ++c) {
            if (c + 1 < NCH) {
                // Issue chunk c+1 into the other buffer while chunk c lands
#pragma unroll
                for (int i = 0; i < 4; ++i) {
                    int o   = (int)threadIdx.x * 16 + i * 4096;
                    int row = o >> 8;
                    int rem = o & 255;
                    cp16_g2l((const char*)hprev + row * (HS * 2)
                                 + (c + 1) * (KC * 2) + rem,
                             (char*)&sh_h[(c + 1) & 1][0][0] + o);
                }
                wait_async_le4();   // chunk c complete, chunk c+1 in flight
            } else {
                wait_async_le0();
            }
            __syncthreads();

            const lds_u16* buf =
                (const lds_u16*)(unsigned int)(size_t)&sh_h[c & 1][0][0];
#pragma unroll
            for (int kk = 0; kk < KC; kk += 32) {
                __builtin_prefetch(Ut + (size_t)bcol * HS + c * KC + kk + 64, 0, 0);
                v16bf a0 = load_a_lds(buf, KC, mh * 32 + r,      kk, half);
                v16bf a1 = load_a_lds(buf, KC, mh * 32 + 16 + r, kk, half);
                v16bf bf = load_b(Ut, HS, bcol, c * KC + kk, half);
                acc0 = wmma_bf16(a0, bf, acc0);
                acc1 = wmma_bf16(a1, bf, acc1);
            }
            __syncthreads();   // reads of buf[c&1] done before it is re-filled
        }

        const float* xwt = xW + (size_t)t * BS * G4;
#pragma unroll
        for (int v = 0; v < 8; ++v) {
            int mA = mh * 32 + half * 8 + v;
            int mB = mA + 16;
            sh_g[q][mA][r] = acc0[v] + xwt[(size_t)mA * G4 + bcol];
            sh_g[q][mB][r] = acc1[v] + xwt[(size_t)mB * G4 + bcol];
        }
        __syncthreads();

        // Cell update for this block's 64x16 (batch x hidden) patch
        unsigned short* hnext = hb + (size_t)((t + 1) & 1) * BS * HS;
        for (int e = threadIdx.x; e < BS * 16; e += 256) {
            int m   = e >> 4;
            int cc  = e & 15;
            int col = h0 + cc;
            float ig = sigm(sh_g[0][m][cc]);
            float fg = sigm(sh_g[1][m][cc]);
            float gg = tanhf(sh_g[2][m][cc]);
            float og = sigm(sh_g[3][m][cc]);
            float cp = cbuf[(size_t)m * HS + col];
            float cn = fg * cp + ig * gg;
            float hn = og * tanhf(cn);
            cbuf[(size_t)m * HS + col]  = cn;
            hnext[(size_t)m * HS + col] = f2bf(hn);
            out[((size_t)m * SEQ + t) * HS + col] = hn;    // hidden_seq [B][S][H]
            if (t == SEQ - 1) {
                size_t tail = (size_t)BS * SEQ * HS;
                out[tail + (size_t)m * HS + col]                   = hn;  // h_t
                out[tail + (size_t)BS * HS + (size_t)m * HS + col] = cn;  // c_t
            }
        }

        // Grid-wide barrier: all blocks must publish h_t before step t+1
        __threadfence();
        __syncthreads();
        if (threadIdx.x == 0) {
            __hip_atomic_fetch_add(&bar[t], 1u, __ATOMIC_RELEASE,
                                   __HIP_MEMORY_SCOPE_AGENT);
            while (__hip_atomic_load(&bar[t], __ATOMIC_ACQUIRE,
                                     __HIP_MEMORY_SCOPE_AGENT) < (unsigned)gridDim.x)
                __builtin_amdgcn_s_sleep(1);
        }
        __syncthreads();
    }
}

// ---------------------------------------------------------------------------
extern "C" void kernel_launch(void* const* d_in, const int* in_sizes, int n_in,
                              void* d_out, int out_size, void* d_ws, size_t ws_size,
                              hipStream_t stream) {
    const float* x    = (const float*)d_in[0];   // [64][512][1024]
    const float* W    = (const float*)d_in[1];   // [1024][4096]
    const float* U    = (const float*)d_in[2];   // [1024][4096]
    const float* bias = (const float*)d_in[3];   // [4096]
    float* out = (float*)d_out;

    char* ws = (char*)d_ws;
    float*          xW  = (float*)(ws + OFF_XW);
    unsigned short* Wt  = (unsigned short*)(ws + OFF_WT);
    unsigned short* Ut  = (unsigned short*)(ws + OFF_UT);
    unsigned short* xb  = (unsigned short*)(ws + OFF_XB);
    unsigned short* hb  = (unsigned short*)(ws + OFF_HB);
    float*          cbf = (float*)(ws + OFF_C);
    unsigned int*   bar = (unsigned int*)(ws + OFF_BAR);

    k_init<<<256, 256, 0, stream>>>(hb, cbf, bar);
    k_f2bf<<<4096, 256, 0, stream>>>(x, xb, (size_t)BS * SEQ * HS);
    k_transpose_bf<<<4096, 256, 0, stream>>>(W, Wt);
    k_transpose_bf<<<4096, 256, 0, stream>>>(U, Ut);
    k_gemm_xw<<<dim3(G4 / 64, (BS * SEQ) / 64), 128, 0, stream>>>(xb, Wt, bias, xW);
    k_recurrent<<<64, 256, 0, stream>>>(Ut, xW, hb, cbf, out, bar);
}
